// UltraMemory_37623913513622
// MI455X (gfx1250) — matrix operand order repaired and checked by reading, not compiled, not run on hardware
//
#include <hip/hip_runtime.h>
#include <hip/hip_bf16.h>
#include <math.h>

typedef __attribute__((ext_vector_type(16))) __bf16 v16bf;
typedef __attribute__((ext_vector_type(8)))  float  v8f;

#define TOK   2048   // B*T
#define DIN   2048
#define QD    512    // HEAD*HALF
#define HALFD 256
#define NKEY  1024   // 2*KEY_NUM per (head, row/col)
#define NSEL  20     // KNN+4
#define NEGINF (-3.402823466e38f)

// ---------------------------------------------------------------------------
// WMMA fragment loaders (contiguous k order k = 16*half + s, 16-bit A 16x32 / B 32x16)
// ---------------------------------------------------------------------------
union Frag { v16bf v; unsigned u[8]; };

// A fragment: lane holds row m, K pairs at {2j} (+16 for upper half-wave)
__device__ __forceinline__ void loadA(const __bf16* rowp, int half, Frag& f) {
#pragma unroll
  for (int j = 0; j < 8; ++j) {
    int koff = (half << 4) | (j << 1);
    f.u[j] = *reinterpret_cast<const unsigned*>(rowp + koff);
  }
}

// B fragment: lane holds col n, K pairs at {2j} (+16 for upper half-wave)
__device__ __forceinline__ void loadB(const __bf16* colp, int half, Frag& f) {
#pragma unroll
  for (int j = 0; j < 8; ++j) {
    int koff = (half << 4) | (j << 1);
    f.u[j] = *reinterpret_cast<const unsigned*>(colp + koff);
  }
}

// ---------------------------------------------------------------------------
// C(MxN) = A(MxK bf16, lda) @ W(NxK bf16, ldw)^T  [+ bias], out fp32 or bf16.
// Wave computes 16x64 of C; block = 8 waves = 32M x 256N.
// All 5 fragment loads of a k-step are issued before the 4 WMMAs so the
// compiler can stage s_wait_loadcnt partial waits (overlap loads with WMMA).
// ---------------------------------------------------------------------------
template <bool OUT_BF16, bool HAS_BIAS>
__global__ __launch_bounds__(256) void gemm_bf16_wmma(
    const __bf16* __restrict__ A, int lda,
    const __bf16* __restrict__ W, int ldw,
    void* __restrict__ Cp, int ldc,
    const float* __restrict__ bias, int K) {
  const int lane = threadIdx.x & 31;
  const int wave = threadIdx.x >> 5;
  const int half = lane >> 4;
  const int r    = lane & 15;
  const int m0 = blockIdx.y * 32 + (wave >> 2) * 16;
  const int n0 = blockIdx.x * 256 + (wave & 3) * 64;

  const __bf16* arow = A + (size_t)(m0 + r) * lda;
  const __bf16* bcol0 = W + (size_t)(n0 + r) * ldw;
  const __bf16* bcol1 = bcol0 + (size_t)16 * ldw;
  const __bf16* bcol2 = bcol0 + (size_t)32 * ldw;
  const __bf16* bcol3 = bcol0 + (size_t)48 * ldw;

  v8f acc[4];
#pragma unroll
  for (int s = 0; s < 4; ++s)
#pragma unroll
    for (int i = 0; i < 8; ++i) acc[s][i] = 0.f;

  for (int k0 = 0; k0 < K; k0 += 32) {
    Frag a, b0, b1, b2, b3;
    loadA(arow + k0, half, a);
    loadB(bcol0 + k0, half, b0);
    loadB(bcol1 + k0, half, b1);
    loadB(bcol2 + k0, half, b2);
    loadB(bcol3 + k0, half, b3);
    if (k0 + 32 < K) __builtin_prefetch(arow + k0 + 32, 0, 0);
    acc[0] = __builtin_amdgcn_wmma_f32_16x16x32_bf16(
        false, a.v, false, b0.v, (short)0, acc[0], false, false);
    acc[1] = __builtin_amdgcn_wmma_f32_16x16x32_bf16(
        false, a.v, false, b1.v, (short)0, acc[1], false, false);
    acc[2] = __builtin_amdgcn_wmma_f32_16x16x32_bf16(
        false, a.v, false, b2.v, (short)0, acc[2], false, false);
    acc[3] = __builtin_amdgcn_wmma_f32_16x16x32_bf16(
        false, a.v, false, b3.v, (short)0, acc[3], false, false);
  }

#pragma unroll
  for (int s = 0; s < 4; ++s)
#pragma unroll
    for (int i = 0; i < 8; ++i) {
      int row = m0 + (half << 3) + i;
      int col = n0 + s * 16 + r;
      float v = acc[s][i];
      if (HAS_BIAS) v += bias[col];
      if (OUT_BF16)
        ((__bf16*)Cp)[(size_t)row * ldc + col] = (__bf16)v;
      else
        ((float*)Cp)[(size_t)row * ldc + col] = v;
    }
}

// ---------------------------------------------------------------------------
// conv3 (causal, depthwise) + cast to bf16
// ---------------------------------------------------------------------------
__global__ void conv_cast_kernel(const float* __restrict__ x,
                                 const float* __restrict__ cw,
                                 const float* __restrict__ cb,
                                 __bf16* __restrict__ y) {
  int idx = blockIdx.x * 256 + threadIdx.x;           // over TOK*DIN
  int d  = idx & (DIN - 1);
  int bt = idx >> 11;
  int t  = bt & 1023;
  size_t base = (size_t)bt * DIN + d;
  float acc = cb[d] + x[base] * cw[d * 3 + 2];
  if (t >= 1) acc += x[base - DIN] * cw[d * 3 + 1];
  if (t >= 2) acc += x[base - 2 * DIN] * cw[d * 3 + 0];
  y[base] = (__bf16)acc;
}

__global__ void cast_f32_bf16(const float* __restrict__ s,
                              __bf16* __restrict__ d, int n) {
  int i = blockIdx.x * 256 + threadIdx.x;
  if (i < n) d[i] = (__bf16)s[i];
}

// WgT[c][l][v*512+d] = valuegroup[v][off_c + d][l]  (off: c==0 -> 512, c==1 -> 0)
__global__ void wgt_kernel(const float* __restrict__ vg, __bf16* __restrict__ wgt) {
  int idx = blockIdx.x * 256 + threadIdx.x;           // over 2*512*2048
  int c   = idx >> 20;
  int rem = idx & ((1 << 20) - 1);
  int l  = rem >> 11;
  int vd = rem & 2047;
  int v  = vd >> 9;
  int d  = vd & 511;
  int off = (c == 0) ? 512 : 0;
  wgt[idx] = (__bf16)vg[((size_t)(v * 1024 + off + d)) * 512 + l];
}

// ---------------------------------------------------------------------------
// Routing: per (token, core) block of 256 threads.
// ---------------------------------------------------------------------------
__global__ __launch_bounds__(256) void routing_kernel(
    const float* __restrict__ srsc,   // planes [rc*2+h][TOK][NKEY]
    const float* __restrict__ core0, const float* __restrict__ core1,
    float* __restrict__ scores, int* __restrict__ rel, int* __restrict__ labels) {
  const int b = blockIdx.x, c = blockIdx.y, tid = threadIdx.x;

  __shared__ float msr[NKEY], msc[NKEY];
  __shared__ float rv[256];
  __shared__ int   ri[256];
  __shared__ float uv[2], tv[2], cm[4];
  __shared__ int   irs[16], ics[16];
  __shared__ float fsr[16][2], fsc[2][16];
  __shared__ float ms[256];
  __shared__ int   sel[NSEL];
  __shared__ float fsco[NSEL];

  if (tid == 0) {
    const float* M = (c == 0) ? core0 : core1;
    float c00 = M[0], c01 = M[1], c10 = M[2], c11 = M[3];
    cm[0] = c00; cm[1] = c01; cm[2] = c10; cm[3] = c11;
    // principal left/right singular vectors of 2x2 M via eig of M*M^T
    float a = c00 * c00 + c01 * c01;
    float bb = c00 * c10 + c01 * c11;
    float cc = c10 * c10 + c11 * c11;
    float phi = 0.5f * atan2f(2.f * bb, a - cc);
    float u0 = cosf(phi), u1 = sinf(phi);
    float v0 = c00 * u0 + c10 * u1;
    float v1 = c01 * u0 + c11 * u1;
    float nrm = sqrtf(v0 * v0 + v1 * v1);
    if (nrm < 1e-20f) { v0 = 1.f; v1 = 0.f; nrm = 1.f; }
    uv[0] = u0; uv[1] = u1; tv[0] = v0 / nrm; tv[1] = v1 / nrm;
  }
  __syncthreads();

  const float* sr0 = srsc + ((size_t)0 * TOK + b) * NKEY;
  const float* sr1 = srsc + ((size_t)1 * TOK + b) * NKEY;
  const float* sc0 = srsc + ((size_t)2 * TOK + b) * NKEY;
  const float* sc1 = srsc + ((size_t)3 * TOK + b) * NKEY;

  for (int k = tid; k < NKEY; k += 256) {
    msr[k] = sr0[k] * uv[0] + sr1[k] * uv[1];
    msc[k] = sc0[k] * tv[0] + sc1[k] * tv[1];
  }
  __syncthreads();

  // top-16 of msr then msc (iterative argmax, tie-break = lowest index)
  for (int pass = 0; pass < 2; ++pass) {
    float* arr = pass ? msc : msr;
    int*   out = pass ? ics : irs;
    for (int round = 0; round < 16; ++round) {
      float best = NEGINF; int bi = NKEY;
      for (int jj = 0; jj < 4; ++jj) {
        int k = tid + jj * 256;
        float v = arr[k];
        if (v > best) { best = v; bi = k; }
      }
      rv[tid] = best; ri[tid] = bi; __syncthreads();
      for (int s = 128; s > 0; s >>= 1) {
        if (tid < s) {
          float v2 = rv[tid + s]; int i2 = ri[tid + s];
          if (v2 > rv[tid] || (v2 == rv[tid] && i2 < ri[tid])) { rv[tid] = v2; ri[tid] = i2; }
        }
        __syncthreads();
      }
      if (tid == 0) { out[round] = ri[0]; arr[ri[0]] = NEGINF; }
      __syncthreads();
    }
  }

  if (tid < 32) {
    int j = tid >> 1, h = tid & 1;
    const float* srh = (h == 0) ? sr0 : sr1;
    const float* sch = (h == 0) ? sc0 : sc1;
    fsr[j][h] = srh[irs[j]];
    fsc[h][j] = sch[ics[j]];
  }
  __syncthreads();

  {
    int j1 = tid >> 4, j2 = tid & 15;
    float a0 = fsr[j1][0], a1 = fsr[j1][1];
    float b0 = fsc[0][j2], b1 = fsc[1][j2];
    ms[tid] = a0 * (cm[0] * b0 + cm[1] * b1) + a1 * (cm[2] * b0 + cm[3] * b1);
  }
  __syncthreads();

  // top-20 of the 256 cross-scores
  for (int round = 0; round < NSEL; ++round) {
    rv[tid] = ms[tid]; ri[tid] = tid; __syncthreads();
    for (int s = 128; s > 0; s >>= 1) {
      if (tid < s) {
        float v2 = rv[tid + s]; int i2 = ri[tid + s];
        if (v2 > rv[tid] || (v2 == rv[tid] && i2 < ri[tid])) { rv[tid] = v2; ri[tid] = i2; }
      }
      __syncthreads();
    }
    if (tid == 0) { sel[round] = ri[0]; fsco[round] = rv[0]; ms[ri[0]] = NEGINF; }
    __syncthreads();
  }

  if (tid < NSEL) {
    int t = sel[tid];
    int j1 = t >> 4, j2 = t & 15;
    int irr = irs[j1], icc = ics[j2];
    // f_ind = irr*1024 + icc ; rel = ((f_ind/512)%512)*512 + f_ind%512
    int rr  = (((irr << 1) + (icc >> 9)) & 511) * 512 + (icc & 511);
    int lab = ((irr >> 9) << 1) + (icc >> 9);
    if (tid >= 16) lab = tid - 16;
    size_t o = ((size_t)c * TOK + b) * NSEL + tid;
    scores[o] = fsco[tid];
    rel[o]    = rr;
    labels[o] = lab;
  }
}

// ---------------------------------------------------------------------------
// agg[c][b][v*512+d] = sum_{k: lab==v} score_k * vb_weight[rel_k][off_c + d]
// ---------------------------------------------------------------------------
__global__ __launch_bounds__(512) void agg_kernel(
    const float* __restrict__ scores, const int* __restrict__ rel,
    const int* __restrict__ labels, const float* __restrict__ vb,
    __bf16* __restrict__ agg) {
  int b = blockIdx.x, c = blockIdx.y, d = threadIdx.x;
  size_t base = ((size_t)c * TOK + b) * NSEL;
  int off = (c == 0) ? 512 : 0;
  float acc[4] = {0.f, 0.f, 0.f, 0.f};
  for (int k = 0; k < NSEL; ++k) {
    int lab = labels[base + k];
    int rr  = rel[base + k];
    float s = scores[base + k];
    float w = vb[(size_t)rr * 1024 + off + d];
    acc[lab] += s * w;
  }
  __bf16* out = agg + ((size_t)c * TOK + b) * 2048;
#pragma unroll
  for (int v = 0; v < 4; ++v) out[v * 512 + d] = (__bf16)acc[v];
}

// ---------------------------------------------------------------------------
extern "C" void kernel_launch(void* const* d_in, const int* in_sizes, int n_in,
                              void* d_out, int out_size, void* d_ws, size_t ws_size,
                              hipStream_t stream) {
  (void)in_sizes; (void)n_in; (void)out_size; (void)ws_size;
  const float* x      = (const float*)d_in[0];
  const float* conv_w = (const float*)d_in[1];
  const float* conv_b = (const float*)d_in[2];
  const float* wq     = (const float*)d_in[3];
  const float* key_p  = (const float*)d_in[4];
  const float* vgrp   = (const float*)d_in[5];
  const float* vb_w   = (const float*)d_in[6];
  const float* core0  = (const float*)d_in[7];
  const float* core1  = (const float*)d_in[8];
  const float* out_w  = (const float*)d_in[9];
  const float* out_b  = (const float*)d_in[10];

  char* ws = (char*)d_ws;
  size_t o = 0;
  __bf16* y_bf    = (__bf16*)(ws + o); o += (size_t)TOK * DIN * 2;      // 8 MB
  __bf16* wq_bf   = (__bf16*)(ws + o); o += (size_t)QD * DIN * 2;       // 2 MB
  __bf16* keys_bf = (__bf16*)(ws + o); o += (size_t)4096 * 256 * 2;     // 2 MB
  __bf16* outw_bf = (__bf16*)(ws + o); o += (size_t)2048 * 1024 * 2;    // 4 MB
  __bf16* q_bf    = (__bf16*)(ws + o); o += (size_t)TOK * QD * 2;       // 2 MB
  float*  srsc    = (float*) (ws + o); o += (size_t)4 * TOK * NKEY * 4; // 32 MB
  float*  scores  = (float*) (ws + o); o += (size_t)2 * TOK * NSEL * 4;
  int*    rel     = (int*)   (ws + o); o += (size_t)2 * TOK * NSEL * 4;
  int*    labels  = (int*)   (ws + o); o += (size_t)2 * TOK * NSEL * 4;
  __bf16* wgt     = (__bf16*)(ws + o); o += (size_t)2 * 512 * 2048 * 2; // 4 MB
  __bf16* agg     = (__bf16*)(ws + o); o += (size_t)2 * TOK * 2048 * 2; // 16 MB
  __bf16* o_cat   = (__bf16*)(ws + o); o += (size_t)TOK * 1024 * 2;     // 4 MB

  // Stage 0: conv + casts
  conv_cast_kernel<<<(TOK * DIN) / 256, 256, 0, stream>>>(x, conv_w, conv_b, y_bf);
  cast_f32_bf16<<<(QD * DIN) / 256, 256, 0, stream>>>(wq, wq_bf, QD * DIN);
  cast_f32_bf16<<<(4096 * 256) / 256, 256, 0, stream>>>(key_p, keys_bf, 4096 * 256);
  cast_f32_bf16<<<(2048 * 1024) / 256, 256, 0, stream>>>(out_w, outw_bf, 2048 * 1024);
  wgt_kernel<<<(2 * 512 * 2048) / 256, 256, 0, stream>>>(vgrp, wgt);

  // Stage 1: q = y @ wq^T  (2048x2048 @ 2048x512), bf16 out
  gemm_bf16_wmma<true, false><<<dim3(QD / 256, TOK / 32), 256, 0, stream>>>(
      y_bf, DIN, wq_bf, DIN, (void*)q_bf, QD, nullptr, DIN);

  // Stage 2: sr/sc planes: for (h, rc): (2048x256) @ (256x1024), f32 out
  for (int h = 0; h < 2; ++h)
    for (int rc = 0; rc < 2; ++rc) {
      const __bf16* Wk = keys_bf + (size_t)(h * 2 + rc) * NKEY * HALFD;
      float* Cp = srsc + (size_t)(rc * 2 + h) * TOK * NKEY;
      gemm_bf16_wmma<false, false><<<dim3(NKEY / 256, TOK / 32), 256, 0, stream>>>(
          q_bf + h * HALFD, QD, Wk, HALFD, (void*)Cp, NKEY, nullptr, HALFD);
    }

  // Stage 3: routing (both cores)
  routing_kernel<<<dim3(TOK, 2), 256, 0, stream>>>(srsc, core0, core1,
                                                   scores, rel, labels);

  // Stage 4: gather + label aggregation
  agg_kernel<<<dim3(TOK, 2), 512, 0, stream>>>(scores, rel, labels, vb_w, agg);

  // Stage 5: o_c = agg_c @ WgT_c^T  -> concat buffer (c=0 cols 0:512, c=1 cols 512:1024)
  for (int c = 0; c < 2; ++c)
    gemm_bf16_wmma<true, false><<<dim3(512 / 256, TOK / 32), 256, 0, stream>>>(
        agg + (size_t)c * TOK * 2048, 2048, wgt + (size_t)c * 512 * 2048, 2048,
        (void*)(o_cat + c * 512), 1024, nullptr, 2048);

  // Stage 6: out = o_cat @ out_w^T + out_b  (2048x1024 @ 1024x2048), f32 out
  gemm_bf16_wmma<false, true><<<dim3(2048 / 256, TOK / 32), 256, 0, stream>>>(
      o_cat, 1024, outw_bf, 1024, d_out, 2048, out_b, 1024);
}
